// Encoder_28741921145341
// MI455X (gfx1250) — matrix operand order repaired and checked by reading, not compiled, not run on hardware
//
#include <hip/hip_runtime.h>

#define B_    128
#define TM1   255
#define N_    256
#define H_    256
#define FOURH 1024
#define KTOT  512          // fused A = [x_tilde_t | h_t] -> K = N_ + H_
#define NBLK  16           // persistent recurrence blocks (16 hidden units each)
#define UPB   16           // hidden units per block
#define MROWS ((size_t)B_ * TM1)   // 32640 flattened (b,t) rows

typedef __attribute__((ext_vector_type(16))) __bf16 v16bf;
typedef __attribute__((ext_vector_type(8)))  float  v8f;

static __device__ __forceinline__ __bf16 f2b(float f) { return (__bf16)f; }
// Fast gate nonlinearities: v_exp_f32 + v_rcp_f32 (TRANS ops co-execute with WMMA).
static __device__ __forceinline__ float sigm(float x) {
  return __builtin_amdgcn_rcpf(1.0f + __expf(-x));
}
static __device__ __forceinline__ float tanh_fast(float x) {
  return __builtin_fmaf(-2.0f, __builtin_amdgcn_rcpf(__expf(2.0f * x) + 1.0f), 1.0f);
}

// ---------------------------------------------------------------------------
// CDNA5 async memory->LDS copy (ASYNCcnt-tracked, no VGPR bounce).
// VDST = LDS byte address (low 32 bits of generic shared pointer),
// VADDR = 64-bit global address.  CDNA5 ISA.
// ---------------------------------------------------------------------------
static __device__ __forceinline__ unsigned lds_addr32(const void* p) {
  return (unsigned)(uintptr_t)p;   // LDS address = low 32 bits of the pointer
}
static __device__ __forceinline__ void async_copy16(unsigned lds_off, const void* gaddr) {
  asm volatile("global_load_async_to_lds_b128 %0, %1, off"
               :
               : "v"(lds_off), "v"((unsigned long long)(uintptr_t)gaddr)
               : "memory");
}
static __device__ __forceinline__ void async_wait0() {
  asm volatile("s_wait_asynccnt 0" ::: "memory");
}

// ---------------------------------------------------------------------------
// Kernel 1: attention logits + softmax over input_size.
// h/c/b_attn terms are constant over n -> they cancel in softmax, so
// alpha[b,n] = softmax_n( sum_t X[b,t,n] * Wa[2H + t] ) is step-invariant.
// ---------------------------------------------------------------------------
__global__ void attn_alpha_kernel(const float* __restrict__ X,
                                  const float* __restrict__ Wa,
                                  float* __restrict__ alpha) {
  __shared__ float wt[TM1];
  __shared__ float red[256];
  const int b = blockIdx.x, n = threadIdx.x;
  if (n < TM1) wt[n] = Wa[2 * H_ + n];
  __syncthreads();

  const float* xp = X + (size_t)b * TM1 * N_ + n;
  float acc = 0.0f;
  for (int t = 0; t < TM1; ++t) acc += xp[(size_t)t * N_] * wt[t];

  red[n] = acc; __syncthreads();
  for (int s = 128; s > 0; s >>= 1) { if (n < s) red[n] = fmaxf(red[n], red[n + s]); __syncthreads(); }
  const float m = red[0]; __syncthreads();
  const float e = __expf(acc - m);
  red[n] = e; __syncthreads();
  for (int s = 128; s > 0; s >>= 1) { if (n < s) red[n] += red[n + s]; __syncthreads(); }
  alpha[b * N_ + n] = e / red[0];
}

// ---------------------------------------------------------------------------
// Kernel 2: X_tilde = alpha * X  (output #1, also LSTM input for all steps)
// ---------------------------------------------------------------------------
__global__ void xtilde_kernel(const float* __restrict__ X,
                              const float* __restrict__ alpha,
                              float* __restrict__ Xt) {
  const size_t i = (size_t)blockIdx.x * 256 + threadIdx.x;   // < B*TM1*N
  const int n = (int)(i & (N_ - 1));
  const size_t b = i / ((size_t)TM1 * N_);
  Xt[i] = X[i] * alpha[b * N_ + n];
}

// ---------------------------------------------------------------------------
// Kernel 3: pack Wc = [W_lstm; U_lstm] to bf16, K-contiguous per column:
// Wc[col][k], k<256 -> W_lstm[k][col], else U_lstm[k-256][col].
// Makes the WMMA B-fragment a single contiguous 32B load per lane.
// ---------------------------------------------------------------------------
__global__ void pack_w_kernel(const float* __restrict__ W,
                              const float* __restrict__ U,
                              unsigned short* __restrict__ Wc_us) {
  __bf16* Wc = (__bf16*)Wc_us;
  const int i = blockIdx.x * 256 + threadIdx.x;              // < 1024*512
  const int col = i >> 9, k = i & (KTOT - 1);
  const float v = (k < N_) ? W[(size_t)k * FOURH + col]
                           : U[(size_t)(k - N_) * FOURH + col];
  Wc[(size_t)col * KTOT + k] = f2b(v);
}

// ---------------------------------------------------------------------------
// Kernel 3b (big-ws path): Z_x = X_tilde_2d @ W_lstm + b_lstm, fully parallel.
// M = B*TM1 = 32640 flattened rows, N = 1024, K = 256.
// Grid (255, 8): 128x128 output tile per block; 8 waves, each 16 rows x 8
// col-tiles; B slice (128 cols x 256 k bf16 = 64KB) async-staged into LDS.
// Bias folded into accumulator init so the recurrence skips it.
// ---------------------------------------------------------------------------
__global__ void __launch_bounds__(256, 1)
zx_gemm_kernel(const float* __restrict__ Xt,                 // (MROWS, 256) f32
               const unsigned short* __restrict__ Wc_us,     // (1024, 512) bf16
               const float* __restrict__ bl,                 // (1024)
               float* __restrict__ Zx) {                     // (MROWS, 1024) f32
  __shared__ __align__(16) __bf16 wcs[128 * N_];             // 64 KB
  const int mb = blockIdx.x * 128;           // row-tile base (flattened b*TM1+t)
  const int nb = blockIdx.y * 128;           // col-tile base
  const int tid = threadIdx.x;
  const int w = tid >> 5, lane = tid & 31;
  const int lhalf = (lane < 16) ? 0 : 1;
  const int l15 = lane & 15;
  const int akb = lhalf * 16, bkb = lhalf * 16;
  const int arow = mb + 16 * w + l15;

  // Async-stage W slice: 128 cols x 512B (k<256, bf16) memory->LDS.
  {
    const unsigned base = lds_addr32(wcs);
    const char* src0 = (const char*)Wc_us;
    for (int q = tid; q < 128 * 32; q += 256) {              // 16B chunks
      const int cl = q >> 5, kc = q & 31;
      async_copy16(base + (unsigned)(cl * 512 + kc * 16),
                   src0 + (size_t)(nb + cl) * (KTOT * 2) + kc * 16);
    }
    async_wait0();
  }
  __syncthreads();

  v8f acc[8];
  #pragma unroll
  for (int n0 = 0; n0 < 8; ++n0) {
    const float bv = bl[nb + n0 * 16 + l15];
    #pragma unroll
    for (int r = 0; r < 8; ++r) acc[n0][r] = bv;
  }

  for (int kk = 0; kk < N_; kk += 32) {
    v16bf a;
    const float* xp = Xt + (size_t)arow * N_ + kk + akb;
    #pragma unroll
    for (int h = 0; h < 8; ++h) a[h] = f2b(xp[h]);
    #pragma unroll
    for (int h = 0; h < 8; ++h) a[8 + h] = f2b(xp[8 + h]);
    #pragma unroll
    for (int n0 = 0; n0 < 8; ++n0) {
      const v16bf bf = *(const v16bf*)(wcs + (n0 * 16 + l15) * N_ + kk + bkb);
      acc[n0] = __builtin_amdgcn_wmma_f32_16x16x32_bf16(
          false, a, false, bf, (short)0, acc[n0], false, false);
    }
  }

  #pragma unroll
  for (int n0 = 0; n0 < 8; ++n0)
    #pragma unroll
    for (int r = 0; r < 8; ++r)
      Zx[(size_t)(mb + 16 * w + r + lhalf * 8) * FOURH + nb + n0 * 16 + l15] = acc[n0][r];
}

// ---------------------------------------------------------------------------
// Software global barrier across NBLK persistent blocks (bar[0]=cnt, bar[1]=gen)
// ---------------------------------------------------------------------------
static __device__ __forceinline__ void gbar(unsigned* bar) {
  __syncthreads();
  if (threadIdx.x == 0) {
    const unsigned gen = __hip_atomic_load(bar + 1, __ATOMIC_ACQUIRE, __HIP_MEMORY_SCOPE_AGENT);
    const unsigned prev = __hip_atomic_fetch_add(bar, 1u, __ATOMIC_ACQ_REL, __HIP_MEMORY_SCOPE_AGENT);
    if (prev + 1 == NBLK) {
      __hip_atomic_store(bar, 0u, __ATOMIC_RELAXED, __HIP_MEMORY_SCOPE_AGENT);
      __hip_atomic_fetch_add(bar + 1, 1u, __ATOMIC_RELEASE, __HIP_MEMORY_SCOPE_AGENT);
    } else {
      while (__hip_atomic_load(bar + 1, __ATOMIC_ACQUIRE, __HIP_MEMORY_SCOPE_AGENT) == gen)
        __builtin_amdgcn_s_sleep(8);
    }
  }
  __syncthreads();
}

// ---------------------------------------------------------------------------
// Kernel 4: persistent LSTM recurrence (templated on Z_x availability).
// Block j owns hidden units [16j,16j+16) -> z columns {g*256 + 16j + u}.
// Wave w owns batch rows [16w,16w+16); 4 accumulators = gates i,f,g,o at the
// same lane positions -> gate fusion is pure in-register VALU.
// c lives in C/D-fragment registers across all 255 steps.
// USE_ZX=true : acc starts from precomputed Z_x(+bias); K loop is h-part only.
// USE_ZX=false: acc starts at 0, bias added at the end; K covers [x_tilde|h].
// ---------------------------------------------------------------------------
template <bool USE_ZX>
__global__ void __launch_bounds__(256, 1)
recurrence_kernel(const float* __restrict__ Xt,             // (B,TM1,N) f32
                  const float* __restrict__ Zx,             // (B,TM1,1024) f32 (zx path)
                  const float* __restrict__ X,              // h0/c0 = X[b,0,0]
                  const unsigned short* __restrict__ Wc_us, // (1024,512) bf16
                  const float* __restrict__ bl,             // (1024)
                  unsigned short* __restrict__ hbuf_us,     // (2,B,H) bf16 double buffer
                  float* __restrict__ Xenc,                 // (B,TM1,H) f32
                  unsigned* __restrict__ bar) {
  __shared__ __align__(16) __bf16 wcs[64 * KTOT];           // 64 KB: block's Wc slice
  __bf16* hbuf = (__bf16*)hbuf_us;

  const int j = blockIdx.x;                  // 0..15
  const int tid = threadIdx.x;
  const int w = tid >> 5;                    // wave -> batch rows [16w,16w+16)
  const int lane = tid & 31;
  const int lhalf = (lane < 16) ? 0 : 1;
  const int l15 = lane & 15;
  const int akb = lhalf * 16, bkb = lhalf * 16;
  const int arow = 16 * w + l15;

  // Async-stage this block's 64 z-columns of Wc (full K=512 -> 1KB per col).
  {
    const unsigned base = lds_addr32(wcs);
    const char* src0 = (const char*)Wc_us;
    for (int q = tid; q < 64 * 64; q += 256) {               // 16B chunks
      const int cl = q >> 6, kc = q & 63;
      const int g = cl >> 4, u = cl & 15;
      async_copy16(base + (unsigned)(cl * 1024 + kc * 16),
                   src0 + (size_t)(g * H_ + j * UPB + u) * (KTOT * 2) + kc * 16);
    }
    async_wait0();
  }
  for (int i = tid; i < B_ * UPB; i += 256) {
    const int b = i >> 4, u = i & 15;
    hbuf[(size_t)b * H_ + j * UPB + u] = f2b(X[(size_t)b * TM1 * N_]);
  }
  v8f cfrag;
  #pragma unroll
  for (int r = 0; r < 8; ++r)
    cfrag[r] = X[(size_t)(16 * w + r + lhalf * 8) * TM1 * N_];

  float bias[4];
  #pragma unroll
  for (int g = 0; g < 4; ++g) bias[g] = bl[g * H_ + j * UPB + l15];

  __syncthreads();
  __threadfence();
  gbar(bar);

  for (int t = 0; t < TM1; ++t) {
    const __bf16* hrd = hbuf + (size_t)(t & 1) * B_ * H_;
    __bf16* hwr = hbuf + (size_t)((t + 1) & 1) * B_ * H_;

    v8f acc[4];
    if constexpr (USE_ZX) {
      if (t + 1 < TM1)
        __builtin_prefetch(Zx + (size_t)arow * TM1 * FOURH + (size_t)(t + 1) * FOURH + j * UPB, 0, 1);
      #pragma unroll
      for (int g = 0; g < 4; ++g) {
        const size_t cb = (size_t)(g * H_ + j * UPB + l15) + (size_t)t * FOURH;
        #pragma unroll
        for (int r = 0; r < 8; ++r)
          acc[g][r] = Zx[(size_t)(16 * w + r + lhalf * 8) * TM1 * FOURH + cb];
      }
    } else {
      if (t + 1 < TM1)
        __builtin_prefetch(Xt + (size_t)arow * TM1 * N_ + (size_t)(t + 1) * N_, 0, 1);
      #pragma unroll
      for (int g = 0; g < 4; ++g)
        #pragma unroll
        for (int r = 0; r < 8; ++r) acc[g][r] = 0.0f;
    }

    const int kk0 = USE_ZX ? N_ : 0;         // zx path: only the h half of K
    for (int kk = kk0; kk < KTOT; kk += 32) {
      v16bf a;
      if (!USE_ZX && kk < N_) {              // x_tilde part: f32 -> bf16 on the fly
        const float* xp = Xt + (size_t)arow * TM1 * N_ + (size_t)t * N_ + kk + akb;
        #pragma unroll
        for (int h = 0; h < 8; ++h) a[h] = f2b(xp[h]);
        #pragma unroll
        for (int h = 0; h < 8; ++h) a[8 + h] = f2b(xp[8 + h]);
      } else {                               // h part: bf16 double buffer
        const __bf16* hp = hrd + (size_t)arow * H_ + (kk - N_) + akb;
        #pragma unroll
        for (int h = 0; h < 8; ++h) a[h] = hp[h];
        #pragma unroll
        for (int h = 0; h < 8; ++h) a[8 + h] = hp[8 + h];
      }
      #pragma unroll
      for (int g = 0; g < 4; ++g) {
        const v16bf bf = *(const v16bf*)(wcs + (g * 16 + l15) * KTOT + kk + bkb);
        acc[g] = __builtin_amdgcn_wmma_f32_16x16x32_bf16(
            false, a, false, bf, (short)0, acc[g], false, false);
      }
    }

    #pragma unroll
    for (int r = 0; r < 8; ++r) {
      float zi = acc[0][r], zf = acc[1][r], zg = acc[2][r], zo = acc[3][r];
      if constexpr (!USE_ZX) { zi += bias[0]; zf += bias[1]; zg += bias[2]; zo += bias[3]; }
      const float ig = sigm(zi);
      const float fg = sigm(zf);
      const float gg = tanh_fast(zg);
      const float og = sigm(zo);
      const float c = fg * cfrag[r] + ig * gg;
      cfrag[r] = c;
      const float h = og * tanh_fast(c);
      const int b = 16 * w + r + lhalf * 8;
      const int ug = j * UPB + l15;
      hwr[(size_t)b * H_ + ug] = f2b(h);
      Xenc[(size_t)b * TM1 * H_ + (size_t)t * H_ + ug] = h;
    }
    __threadfence();
    gbar(bar);
  }
}

// ---------------------------------------------------------------------------
extern "C" void kernel_launch(void* const* d_in, const int* in_sizes, int n_in,
                              void* d_out, int out_size, void* d_ws, size_t ws_size,
                              hipStream_t stream) {
  (void)in_sizes; (void)n_in; (void)out_size;
  const float* X  = (const float*)d_in[0];
  const float* Wa = (const float*)d_in[1];
  const float* Wl = (const float*)d_in[3];
  const float* Ul = (const float*)d_in[4];
  const float* bl = (const float*)d_in[5];

  char* ws = (char*)d_ws;
  float* alpha          = (float*)ws;                                 // 128 KB
  unsigned short* Wc    = (unsigned short*)(ws + 131072);             // 1 MB
  unsigned short* hbuf  = (unsigned short*)(ws + 131072 + 1048576);   // 128 KB
  unsigned* bar         = (unsigned*)(ws + 131072 + 1048576 + 131072);
  const size_t zx_off   = 131072 + 1048576 + 131072 + 256;            // 256B-aligned
  float* Zx             = (float*)(ws + zx_off);
  const size_t zx_bytes = MROWS * FOURH * sizeof(float);              // 133.7 MB
  const bool use_zx     = ws_size >= zx_off + zx_bytes;

  float* Xt   = (float*)d_out;                                        // (B,TM1,N)
  float* Xenc = (float*)d_out + MROWS * N_;                           // (B,TM1,H)

  hipMemsetAsync(bar, 0, 2 * sizeof(unsigned), stream);
  attn_alpha_kernel<<<B_, 256, 0, stream>>>(X, Wa, alpha);
  xtilde_kernel<<<(int)(MROWS * N_ / 256), 256, 0, stream>>>(X, alpha, Xt);
  pack_w_kernel<<<(FOURH * KTOT) / 256, 256, 0, stream>>>(Wl, Ul, Wc);

  if (use_zx) {
    zx_gemm_kernel<<<dim3(MROWS / 128, FOURH / 128), 256, 0, stream>>>(Xt, Wc, bl, Zx);
    recurrence_kernel<true><<<NBLK, 256, 0, stream>>>(Xt, Zx, X, Wc, bl, hbuf, Xenc, bar);
  } else {
    recurrence_kernel<false><<<NBLK, 256, 0, stream>>>(Xt, nullptr, X, Wc, bl, hbuf, Xenc, bar);
  }
}
